// OuroborosMoE_43430709297943
// MI455X (gfx1250) — compile-verified, matrix operand order checked by reading, not run on hardware
//
#include <hip/hip_runtime.h>
#include <hip/hip_bf16.h>
#include <math.h>

// ---------------------------------------------------------------------------
// OuroborosMoE forward for MI455X (gfx1250): gathered top-2 MoE MLP with
// bf16 WMMA (v_wmma_f32_16x16x32_bf16) GEMMs, fused GELU, scatter-add combine.
// M-block = 64 rows (16 waves) to halve per-expert weight re-streaming from L2.
// ---------------------------------------------------------------------------

#define D_MODEL 512
#define H_DIM   2048
#define NEXP    8
#define TOPK    2
#define NTOK    (8 * 4096)          // 32768 tokens
#define NPAIRS  (NTOK * TOPK)       // 65536 token-expert pairs
#define M_BLK   64                  // pair rows per workgroup
#define NTHREADS 512                // 16 waves: 4(M) x 4(N)
#define HC      64                  // hidden-dim chunk per iteration
#define KT1     (D_MODEL / 32)      // 16  k-tiles in GEMM1
#define NT1     (H_DIM / 16)        // 128 n-tiles in GEMM1
#define KT2     (H_DIM / 32)        // 64  k-tiles in GEMM2
#define NT2     (D_MODEL / 16)      // 32  n-tiles in GEMM2
#define TILE_HALFS 512              // one packed 32x16 bf16 B tile = 1KB
#define LDX     (D_MODEL + 8)       // padded LDS stride (bank-conflict dodge)
#define LDH     (HC + 8)
#define MAXBLOCKS (NEXP + NPAIRS / M_BLK)   // 1032

typedef __attribute__((ext_vector_type(16))) __bf16 v16bf;
typedef __attribute__((ext_vector_type(8)))  float  v8f;

static __device__ inline unsigned short f2bf_bits(float f) {
  union { float f; unsigned u; } a; a.f = f;
  unsigned u = a.u;
  u += 0x7FFFu + ((u >> 16) & 1u);        // round-to-nearest-even
  return (unsigned short)(u >> 16);
}
static __device__ inline __bf16 f2bf(float f) {
  union { unsigned short s; __bf16 b; } r; r.s = f2bf_bits(f);
  return r.b;
}
static __device__ inline float gelu_exact(float v) {
  return 0.5f * v * (1.0f + erff(v * 0.70710678118654752f));
}

// A fragment (16x32 bf16, M x K) from row-major LDS tile.
// lane 0-15: row M=lane,    K = {k0+0..7,  k0+16..23}
// lane16-31: row M=lane-16, K = {k0+8..15, k0+24..31}
static __device__ inline v16bf loadA(const __bf16* base, int ld, int lane, int k0) {
  int row = lane & 15;
  int kb  = (lane >> 4) * 8;
  const __bf16* p = base + (size_t)row * ld + k0 + kb;
  union { v16bf v; uint4 q[2]; } u;
  u.q[0] = *(const uint4*)(p);
  u.q[1] = *(const uint4*)(p + 16);
  return u.v;
}

// B fragment: packed tile, lane l's 16 halfs are contiguous at l*16.
static __device__ inline v16bf loadB(const __bf16* tile, int lane) {
  const uint4* p = (const uint4*)(tile + (size_t)lane * 16);
  union { v16bf v; uint4 q[2]; } u;
  u.q[0] = p[0];
  u.q[1] = p[1];
  return u.v;
}

// ---------------------------------------------------------------------------
// Kernel 1: convert W1/W2 f32 -> bf16, pre-swizzled into WMMA B-fragment
// layout (32x16 K-major tiles; lane l owns 16 contiguous halfs).
// tileId < nW1 : W1 tiles, layout [e][nt][kt]
// else         : W2 tiles, layout [e][kt][nt]
// ---------------------------------------------------------------------------
__global__ __launch_bounds__(256)
void pack_weights(const float* __restrict__ W1, const float* __restrict__ W2,
                  __bf16* __restrict__ W1p, __bf16* __restrict__ W2p) {
  const int tileId = blockIdx.x * 8 + (threadIdx.x >> 5);
  const int lane   = threadIdx.x & 31;
  const int nW1 = NEXP * NT1 * KT1;       // 16384
  const int nW2 = NEXP * KT2 * NT2;       // 16384

  union { uint4 q[2]; unsigned short s[16]; } u;
  const int nloc = lane & 15;
  const int kloc = (lane >> 4) * 8;

  if (tileId < nW1) {
    int e   = tileId / (NT1 * KT1);
    int rem = tileId % (NT1 * KT1);
    int nt  = rem / KT1;
    int kt  = rem % KT1;
    const float* src = W1 + (size_t)e * D_MODEL * H_DIM;   // [d][h] row-major
    int n  = nt * 16 + nloc;
    int kb = kt * 32 + kloc;
#pragma unroll
    for (int i = 0; i < 8; ++i) {
      u.s[i]     = f2bf_bits(src[(size_t)(kb + i)      * H_DIM + n]);
      u.s[8 + i] = f2bf_bits(src[(size_t)(kb + 16 + i) * H_DIM + n]);
    }
    uint4* dst = (uint4*)(W1p + (size_t)tileId * TILE_HALFS + (size_t)lane * 16);
    dst[0] = u.q[0]; dst[1] = u.q[1];
  } else if (tileId < nW1 + nW2) {
    int t2  = tileId - nW1;
    int e   = t2 / (KT2 * NT2);
    int rem = t2 % (KT2 * NT2);
    int kt  = rem / NT2;
    int nt  = rem % NT2;
    const float* src = W2 + (size_t)e * H_DIM * D_MODEL;   // [h][d] row-major
    int n  = nt * 16 + nloc;
    int kb = kt * 32 + kloc;
#pragma unroll
    for (int i = 0; i < 8; ++i) {
      u.s[i]     = f2bf_bits(src[(size_t)(kb + i)      * D_MODEL + n]);
      u.s[8 + i] = f2bf_bits(src[(size_t)(kb + 16 + i) * D_MODEL + n]);
    }
    uint4* dst = (uint4*)(W2p + (size_t)t2 * TILE_HALFS + (size_t)lane * 16);
    dst[0] = u.q[0]; dst[1] = u.q[1];
  }
}

// ---------------------------------------------------------------------------
// Routing: histogram -> scan -> scatter into per-expert compacted lists.
// ---------------------------------------------------------------------------
__global__ __launch_bounds__(256)
void route_count(const int* __restrict__ idx, int* __restrict__ counts) {
  int p = blockIdx.x * 256 + threadIdx.x;
  if (p < NPAIRS) atomicAdd(&counts[idx[p]], 1);
}

__global__ void route_scan(const int* __restrict__ counts,
                           int* __restrict__ offsets,
                           int* __restrict__ blockStart,
                           int* __restrict__ cursor) {
  if (threadIdx.x == 0) {
    int off = 0, bs = 0;
    for (int e = 0; e < NEXP; ++e) {
      offsets[e]    = off;
      blockStart[e] = bs;
      cursor[e]     = off;
      off += counts[e];
      bs  += (counts[e] + M_BLK - 1) / M_BLK;
    }
    offsets[NEXP]    = off;
    blockStart[NEXP] = bs;
  }
}

__global__ __launch_bounds__(256)
void route_scatter(const int* __restrict__ idx, const float* __restrict__ w,
                   int* __restrict__ cursor,
                   int* __restrict__ gtok, float* __restrict__ gw) {
  int p = blockIdx.x * 256 + threadIdx.x;
  if (p < NPAIRS) {
    int e    = idx[p];
    int slot = atomicAdd(&cursor[e], 1);
    gtok[slot] = p / TOPK;          // token id
    gw[slot]   = w[p];              // combine weight
  }
}

// out = x  (residual base; expert contributions are atomically accumulated)
__global__ __launch_bounds__(256)
void init_out(const float4* __restrict__ x4, float4* __restrict__ out4, int n4) {
  int i = blockIdx.x * 256 + threadIdx.x;
  if (i < n4) out4[i] = x4[i];
}

// ---------------------------------------------------------------------------
// Fused MoE MLP block: 64 gathered pair-rows x full d=512 output.
// 16 waves arranged 4(M) x 4(N).  Per 64-wide hidden chunk:
//   GEMM1 (Xs 64x512 . W1 512x64) -> GELU+b1 -> Hs bf16 (LDS)
//   GEMM2 (Hs 64x64 . W2 64x512)  -> accumulate Y (8 x v8f per wave)
// Epilogue: Y + b2, scale by per-row combine weight, atomic-add into out.
// ---------------------------------------------------------------------------
__global__ __launch_bounds__(NTHREADS)
void moe_gemm(const float* __restrict__ x,
              const float* __restrict__ b1, const float* __restrict__ b2,
              const __bf16* __restrict__ W1p, const __bf16* __restrict__ W2p,
              const int* __restrict__ offsets, const int* __restrict__ blockStart,
              const int* __restrict__ gtok, const float* __restrict__ gw,
              float* __restrict__ out) {
  __shared__ __bf16 Xs[M_BLK][LDX];    // ~65 KB
  __shared__ __bf16 Hs[M_BLK][LDH];    // ~9 KB
  __shared__ int    toks[M_BLK];
  __shared__ float  wrow[M_BLK];

  const int b = blockIdx.x;
  if (b >= blockStart[NEXP]) return;        // uniform early-exit
  int e = 0;
  while (b >= blockStart[e + 1]) ++e;       // <= 8 iterations

  const int rowBeg = offsets[e] + (b - blockStart[e]) * M_BLK;
  const int rowEnd = offsets[e + 1];

  const int tid  = threadIdx.x;
  const int lane = tid & 31;
  const int wid  = tid >> 5;
  const int wm   = wid & 3;                 // M-tile (0..3)
  const int wn   = wid >> 2;                // N-group (0..3), 128 cols each

  if (tid < M_BLK) {
    int r = rowBeg + tid;
    bool valid = r < rowEnd;
    toks[tid] = valid ? gtok[r] : -1;
    wrow[tid] = valid ? gw[r]   : 0.0f;
  }
  __syncthreads();

  // Stage gathered X rows into LDS as bf16 (padded rows = 0).
  for (int i = tid; i < M_BLK * D_MODEL; i += NTHREADS) {
    int r = i >> 9;                         // /512
    int c = i & (D_MODEL - 1);
    int t = toks[r];
    float f = (t >= 0) ? x[(size_t)t * D_MODEL + c] : 0.0f;
    Xs[r][c] = f2bf(f);
  }
  __syncthreads();

  v8f acc[8];
#pragma unroll
  for (int j = 0; j < 8; ++j) acc[j] = (v8f){0, 0, 0, 0, 0, 0, 0, 0};

  const __bf16* W1e = W1p + (size_t)e * NT1 * KT1 * TILE_HALFS;
  const __bf16* W2e = W2p + (size_t)e * KT2 * NT2 * TILE_HALFS;

  const int rhalf = (lane >> 4) * 8;        // C/D layout: M = r + 8*(lane>=16)
  const int col   = lane & 15;              //              N = lane & 15

  for (int ch = 0; ch < H_DIM / HC; ++ch) {
    // Prefetch next chunk's B-tile streams into L2/L0 while we compute.
    if (ch + 1 < H_DIM / HC) {
      const __bf16* w1n = W1e + ((size_t)((ch + 1) * 4 + wn) * KT1) * TILE_HALFS;
      const __bf16* w2n = W2e + ((size_t)((ch + 1) * (HC / 32)) * NT2 + wn * 8) * TILE_HALFS;
      __builtin_prefetch(w1n + (size_t)lane * 16, 0, 1);
      __builtin_prefetch(w2n + (size_t)lane * 16, 0, 1);
    }

    // ---- GEMM1: this wave's 16x16 H tile (rows wm*16.., cols ch*64+wn*16..)
    v8f hacc = (v8f){0, 0, 0, 0, 0, 0, 0, 0};
    const __bf16* w1t = W1e + ((size_t)(ch * 4 + wn) * KT1) * TILE_HALFS;
#pragma unroll 4
    for (int kt = 0; kt < KT1; ++kt) {
      v16bf a  = loadA(&Xs[wm * 16][0], LDX, lane, kt * 32);
      v16bf bf = loadB(w1t + (size_t)kt * TILE_HALFS, lane);
      hacc = __builtin_amdgcn_wmma_f32_16x16x32_bf16(
          false, a, false, bf, (short)0, hacc, false, false);
    }
    // GELU(+b1) epilogue -> bf16 chunk in LDS
    {
      float b1v = b1[e * H_DIM + ch * HC + wn * 16 + col];
#pragma unroll
      for (int r = 0; r < 8; ++r) {
        float v = gelu_exact(hacc[r] + b1v);
        Hs[wm * 16 + rhalf + r][wn * 16 + col] = f2bf(v);
      }
    }
    __syncthreads();

    // ---- GEMM2: Y += Hs(64x64) . W2[chunk](64x512)
#pragma unroll
    for (int kk = 0; kk < HC / 32; ++kk) {
      v16bf a = loadA(&Hs[wm * 16][0], LDH, lane, kk * 32);
      int kt2 = ch * (HC / 32) + kk;
      const __bf16* w2t = W2e + ((size_t)kt2 * NT2 + wn * 8) * TILE_HALFS;
#pragma unroll
      for (int j = 0; j < 8; ++j) {
        v16bf bf = loadB(w2t + (size_t)j * TILE_HALFS, lane);
        acc[j] = __builtin_amdgcn_wmma_f32_16x16x32_bf16(
            false, a, false, bf, (short)0, acc[j], false, false);
      }
    }
    __syncthreads();   // Hs reuse next chunk
  }

  // Epilogue: + b2, scale by combine weight, scatter-add into out.
#pragma unroll
  for (int j = 0; j < 8; ++j) {
    int n = wn * 128 + j * 16 + col;
    float b2v = b2[e * D_MODEL + n];
#pragma unroll
    for (int r = 0; r < 8; ++r) {
      int lrow = wm * 16 + rhalf + r;
      int t = toks[lrow];
      if (t >= 0) {
        float val = (acc[j][r] + b2v) * wrow[lrow];
        atomicAdd(&out[(size_t)t * D_MODEL + n], val);
      }
    }
  }
}

// ---------------------------------------------------------------------------
// Launch
// ---------------------------------------------------------------------------
extern "C" void kernel_launch(void* const* d_in, const int* in_sizes, int n_in,
                              void* d_out, int out_size, void* d_ws, size_t ws_size,
                              hipStream_t stream) {
  const float* x   = (const float*)d_in[0];   // [B,T,d]
  const float* ew  = (const float*)d_in[1];   // [B,T,K]
  const float* W1  = (const float*)d_in[2];   // [E,d,h]
  const float* b1  = (const float*)d_in[3];   // [E,h]
  const float* W2  = (const float*)d_in[4];   // [E,h,d]
  const float* b2  = (const float*)d_in[5];   // [E,d]
  const int*  eidx = (const int*)d_in[6];     // [B,T,K]
  float* out = (float*)d_out;

  // Workspace carve-up (~33 MB total)
  char* ws = (char*)d_ws;
  const size_t W1P_BYTES = (size_t)NEXP * NT1 * KT1 * TILE_HALFS * 2;  // 16 MB
  const size_t W2P_BYTES = (size_t)NEXP * KT2 * NT2 * TILE_HALFS * 2;  // 16 MB
  __bf16* W1p = (__bf16*)ws;
  __bf16* W2p = (__bf16*)(ws + W1P_BYTES);
  char* meta  = ws + W1P_BYTES + W2P_BYTES;
  int* counts     = (int*)meta;                 // 8
  int* offsets    = counts + 8;                 // 9
  int* blockStart = offsets + 16;               // 9 (padded)
  int* cursor     = blockStart + 16;            // 8
  int* gtok       = (int*)(meta + 4096);        // NPAIRS
  float* gw       = (float*)(meta + 4096 + (size_t)NPAIRS * 4);

  // 1) pack/convert weights into WMMA fragment layout (bf16)
  pack_weights<<<4096, 256, 0, stream>>>(W1, W2, W1p, W2p);

  // 2) routing
  hipMemsetAsync(counts, 0, NEXP * sizeof(int), stream);
  route_count<<<(NPAIRS + 255) / 256, 256, 0, stream>>>(eidx, counts);
  route_scan<<<1, 32, 0, stream>>>(counts, offsets, blockStart, cursor);
  route_scatter<<<(NPAIRS + 255) / 256, 256, 0, stream>>>(eidx, ew, cursor, gtok, gw);

  // 3) out = x (residual base)
  {
    int n4 = NTOK * D_MODEL / 4;
    init_out<<<(n4 + 255) / 256, 256, 0, stream>>>((const float4*)x, (float4*)out, n4);
  }

  // 4) fused gathered MoE MLP (WMMA)
  moe_gemm<<<MAXBLOCKS, NTHREADS, 0, stream>>>(x, b1, b2, W1p, W2p,
                                               offsets, blockStart, gtok, gw, out);
}
